// DAWN_57423712747577
// MI455X (gfx1250) — matrix operand order repaired and checked by reading, not compiled, not run on hardware
//
#include <hip/hip_runtime.h>
#include <hip/hip_bf16.h>
#include <stdint.h>

// ---------------------------------------------------------------------------
// B=4, S=1024, D=1024, DB=128, R=128, N_QK=N_V=2048, N_KNOW=4096,
// MB_QK=MB_V=16, MB_KNOW=32, H=16, dh=64. Tokens T = 4096.
// All GEMM dims are multiples of 64 -> no edge guards.
//
// B-operands of every GEMM are pre-packed to "K-pair u32" format:
//   PB[k/2][n] = (bf16(B[k+1][n]) << 16) | bf16(B[k][n])
// so each WMMA B-fragment VGPR is one ds_load_b32, and each A-fragment is
// two ds_load_b128 (CDNA5 ISA 7.12.2 16-bit layouts).
// ---------------------------------------------------------------------------

typedef unsigned short u16;
typedef __attribute__((ext_vector_type(16))) __bf16 v16bf;
typedef __attribute__((ext_vector_type(8)))  float  v8f;

struct __attribute__((aligned(16))) U4 { unsigned a, b, c, d; };

union FragBF {
  v16bf h;
  unsigned u[8];
  U4 q[2];
};

static __device__ __forceinline__ u16 f2bf(float f) {
  unsigned u = __float_as_uint(f);
  u = u + 0x7FFFu + ((u >> 16) & 1u);   // round-to-nearest-even
  return (u16)(u >> 16);
}
static __device__ __forceinline__ float bf2f(u16 u) {
  return __uint_as_float(((unsigned)u) << 16);
}
static __device__ __forceinline__ unsigned pack2(float lo, float hi) {
  return ((unsigned)f2bf(hi) << 16) | (unsigned)f2bf(lo);
}
// scale both bf16 halves of a u32 by s
static __device__ __forceinline__ unsigned scale_pk(unsigned p, float s) {
  float lo = bf2f((u16)(p & 0xffffu)) * s;
  float hi = bf2f((u16)(p >> 16)) * s;
  return pack2(lo, hi);
}
// interleave two bf16-pair words (same two dims, adjacent keys) into k-pairs
static __device__ __forceinline__ void interleave(unsigned a, unsigned b,
                                                  unsigned& lo, unsigned& hi) {
  lo = ((b & 0xffffu) << 16) | (a & 0xffffu);
  hi = (b & 0xffff0000u) | (a >> 16);
}
static __device__ __forceinline__ v8f wmma_bf16(v16bf a, v16bf b, v8f c) {
  return __builtin_amdgcn_wmma_f32_16x16x32_bf16(false, a, false, b,
                                                 (short)0, c, false, false);
}

// ---------------------------------------------------------------------------
// Elementwise / small kernels
// ---------------------------------------------------------------------------
__global__ void k_copy(float* __restrict__ dst, const float* __restrict__ src, int n) {
  for (int i = blockIdx.x * blockDim.x + threadIdx.x; i < n; i += gridDim.x * blockDim.x)
    dst[i] = src[i];
}
__global__ void k_zero(float* __restrict__ dst, int n) {
  for (int i = blockIdx.x * blockDim.x + threadIdx.x; i < n; i += gridDim.x * blockDim.x)
    dst[i] = 0.f;
}
// plain f32 -> bf16 (A-operand activations)
__global__ void k_conv(u16* __restrict__ dst, const float* __restrict__ src, int n) {
  for (int i = blockIdx.x * blockDim.x + threadIdx.x; i < n; i += gridDim.x * blockDim.x)
    dst[i] = f2bf(src[i]);
}
// strided f32 -> bf16 (slice columns out of a wider row)
__global__ void k_convS(u16* __restrict__ dst, const float* __restrict__ src,
                        int rows, int cols, int stride, int off) {
  int total = rows * cols;
  for (int i = blockIdx.x * blockDim.x + threadIdx.x; i < total; i += gridDim.x * blockDim.x) {
    int t = i / cols, c = i - t * cols;
    dst[i] = f2bf(src[(size_t)t * stride + off + c]);
  }
}
// row-major f32 [K][N] -> packed-B u32 [K/2][N]
__global__ void k_convPB(unsigned* __restrict__ dst, const float* __restrict__ src,
                         int K, int N) {
  int total = (K >> 1) * N;
  for (int i = blockIdx.x * blockDim.x + threadIdx.x; i < total; i += gridDim.x * blockDim.x) {
    int kp = i / N, n = i - kp * N;
    dst[i] = pack2(src[(size_t)(2 * kp) * N + n], src[(size_t)(2 * kp + 1) * N + n]);
  }
}
// basis weights [NB][1024][128] -> packed-B u32 [512][NB*128] (k = d)
__global__ void k_packFP(unsigned* __restrict__ dst, const float* __restrict__ src,
                         int NB, int total) {
  int NN = NB * 128;
  for (int i = blockIdx.x * blockDim.x + threadIdx.x; i < total; i += gridDim.x * blockDim.x) {
    int kp = i / NN, col = i - kp * NN;
    int n = col >> 7, r = col & 127;
    const float* base = src + (size_t)n * 1024 * 128 + (size_t)(2 * kp) * 128 + r;
    dst[i] = pack2(base[0], base[128]);
  }
}
// normalize embedding rows [N][128], transpose, pack -> u32 [64][N]
__global__ __launch_bounds__(128)
void k_embTP(const float* __restrict__ emb, unsigned* __restrict__ dst, int N) {
  int n = blockIdx.x, d = threadIdx.x;
  float v = emb[(size_t)n * 128 + d];
  __shared__ float red[128];
  __shared__ float sv[128];
  red[d] = v * v; sv[d] = v; __syncthreads();
  for (int o = 64; o > 0; o >>= 1) { if (d < o) red[d] += red[d + o]; __syncthreads(); }
  float inv = 1.f / (sqrtf(red[0]) + 1e-8f);
  if (d < 64)
    dst[(size_t)d * N + n] = pack2(sv[2 * d] * inv, sv[2 * d + 1] * inv);
}

// LayerNorm: one block per token row of D, writes f32 + bf16
__global__ __launch_bounds__(256)
void k_ln(const float* __restrict__ x, const float* __restrict__ sc,
          const float* __restrict__ bi, float* __restrict__ of,
          u16* __restrict__ ob, int D) {
  int t = blockIdx.x, tid = threadIdx.x;
  const float* row = x + (size_t)t * D;
  __shared__ float red[256];
  float s = 0.f;
  for (int d = tid; d < D; d += 256) s += row[d];
  red[tid] = s; __syncthreads();
  for (int o = 128; o > 0; o >>= 1) { if (tid < o) red[tid] += red[tid + o]; __syncthreads(); }
  float mean = red[0] / (float)D;
  __syncthreads();
  float v = 0.f;
  for (int d = tid; d < D; d += 256) { float z = row[d] - mean; v += z * z; }
  red[tid] = v; __syncthreads();
  for (int o = 128; o > 0; o >>= 1) { if (tid < o) red[tid] += red[tid + o]; __syncthreads(); }
  float inv = rsqrtf(red[0] / (float)D + 1e-6f);
  for (int d = tid; d < D; d += 256) {
    float y = (row[d] - mean) * inv * sc[d] + bi[d];
    of[(size_t)t * D + d] = y;
    ob[(size_t)t * D + d] = f2bf(y);
  }
}

// small GEMV: tau[t, 0..C) = n1[t,:] @ K[D,C] + b   (C <= 3)
__global__ __launch_bounds__(256)
void k_tau(const float* __restrict__ n1, const float* __restrict__ Km,
           const float* __restrict__ bias, float* __restrict__ out, int D, int C) {
  int t = blockIdx.x, tid = threadIdx.x;
  __shared__ float red[3 * 256];
  float part[3] = {0.f, 0.f, 0.f};
  for (int d = tid; d < D; d += 256) {
    float xv = n1[(size_t)t * D + d];
    for (int c = 0; c < C; ++c) part[c] += xv * Km[d * C + c];
  }
  for (int c = 0; c < C; ++c) red[c * 256 + tid] = part[c];
  __syncthreads();
  for (int o = 128; o > 0; o >>= 1) {
    if (tid < o) for (int c = 0; c < C; ++c) red[c * 256 + tid] += red[c * 256 + tid + o];
    __syncthreads();
  }
  if (tid < C) out[(size_t)t * C + tid] = red[tid * 256] + bias[tid];
}

// threshold gate + per-neuron mean accumulation + group gate
__global__ __launch_bounds__(256)
void k_gate(float* __restrict__ scores, const float* __restrict__ tau,
            int tauStride, int tauOff, int N, int ngroups,
            float* __restrict__ nsum, float* __restrict__ gb) {
  int t = blockIdx.x, tid = threadIdx.x;
  float tv = tau[(size_t)t * tauStride + tauOff];
  float* row = scores + (size_t)t * N;
  __shared__ float red[256];
  __shared__ float gs[32];
  __shared__ float sh[2];
  if (tid < 32) gs[tid] = 0.f;
  float s = 0.f;
  for (int i = tid; i < N; i += 256) {
    float g = row[i] - tv; g = g > 0.f ? g : 0.f;
    row[i] = g; s += g;
  }
  red[tid] = s; __syncthreads();
  for (int o = 128; o > 0; o >>= 1) { if (tid < o) red[tid] += red[tid + o]; __syncthreads(); }
  if (tid == 0) sh[0] = 1.f / (red[0] + 1e-8f);
  __syncthreads();
  float inv = sh[0];
  int gsize = N / ngroups;
  for (int i = tid; i < N; i += 256) {
    float g = row[i] * inv;
    atomicAdd(&nsum[i], g);
    atomicAdd(&gs[i / gsize], g);
  }
  __syncthreads();
  if (tid == 0) {
    float tot = 0.f;
    for (int j = 0; j < ngroups; ++j) tot += gs[j];
    sh[1] = 1.f / (tot + 1e-8f);
  }
  __syncthreads();
  if (tid < ngroups) gb[(size_t)t * ngroups + tid] = gs[tid] * sh[1];
}

// h[t,r] = sum_n w[t,n] * all_h[t, n*128 + r]
__global__ __launch_bounds__(128)
void k_wred(const float* __restrict__ all_h, const float* __restrict__ w,
            float* __restrict__ h, int NB) {
  int t = blockIdx.x, r = threadIdx.x;
  const float* a = all_h + (size_t)t * NB * 128;
  const float* wt = w + (size_t)t * NB;
  float s = 0.f;
  for (int n = 0; n < NB; ++n) s += wt[n] * a[n * 128 + r];
  h[(size_t)t * 128 + r] = s;
}

// expanded bf16 operand for restore GEMM: dst[t, n*128+r] = w[t,n]*h[t,r]
__global__ void k_exp(u16* __restrict__ dst, const float* __restrict__ h,
                      const float* __restrict__ w, int NB, int total) {
  int nbr = NB * 128;
  for (int idx = blockIdx.x * blockDim.x + threadIdx.x; idx < total;
       idx += gridDim.x * blockDim.x) {
    int t = idx / nbr;
    int rem = idx - t * nbr;
    int n = rem >> 7, r = rem & 127;
    dst[idx] = f2bf(w[(size_t)t * NB + n] * h[(size_t)t * 128 + r]);
  }
}

// aux loss reduction
__global__ __launch_bounds__(256)
void k_aux(const float* __restrict__ sq, const float* __restrict__ sk,
           const float* __restrict__ sv, const float* __restrict__ skn,
           float inv_ntok, float* __restrict__ dst) {
  __shared__ float red[256];
  int tid = threadIdx.x;
  float a = 0.f;
  const float tq = 1.f / 2048.f, tk2 = 1.f / 4096.f;
  for (int i = tid; i < 2048; i += 256) {
    float d1 = sq[i] * inv_ntok - tq; a += d1 * d1 * 2048.f;
    float d2 = sk[i] * inv_ntok - tq; a += d2 * d2 * 2048.f;
    float d3 = sv[i] * inv_ntok - tq; a += d3 * d3 * 2048.f;
  }
  for (int i = tid; i < 4096; i += 256) {
    float d = skn[i] * inv_ntok - tk2; a += d * d * 4096.f;
  }
  red[tid] = a; __syncthreads();
  for (int o = 128; o > 0; o >>= 1) { if (tid < o) red[tid] += red[tid + o]; __syncthreads(); }
  if (tid == 0) *dst = red[0];
}

// ---------------------------------------------------------------------------
// WMMA GEMM: C[M,N] = A_bf16[M,K] @ PB[K/2,N] (+bias)
// 256 threads = 8 waves; 64x64 tile; K chunk 64 (4 wmma per barrier pair).
// flags: 1 = store bf16, 2 = accumulate into f32 C, else plain f32 store.
// ---------------------------------------------------------------------------
__global__ __launch_bounds__(256)
void k_gemm(const u16* __restrict__ A, const unsigned* __restrict__ PB, void* Cv,
            const float* __restrict__ bias, int M, int N, int K, int flags) {
  __shared__ u16 As[64][72];        // 64x64 bf16 tile   (144B rows)
  __shared__ unsigned Bs[32][68];   // 32kp x 64n u32    (272B rows)
  int m0 = blockIdx.y * 64, n0 = blockIdx.x * 64;
  int tid = threadIdx.x, lane = tid & 31, wave = tid >> 5;
  int mw = wave & 3, nw = wave >> 2;
  int hi = lane >> 4, li = lane & 15;
  v8f acc0, acc1;
  #pragma unroll
  for (int v = 0; v < 8; ++v) { acc0[v] = 0.f; acc1[v] = 0.f; }

  int ra = tid >> 2, ca = (tid & 3) * 16;   // A: 64 rows x 4 thr x 16 elems
  int rb = tid >> 3, cb = (tid & 7) * 8;    // B: 32 kp  x 8 thr x 8 u32
  int am = mw * 16 + li;
  int bn = nw * 32 + li;
  for (int k0 = 0; k0 < K; k0 += 64) {
    const u16* asrc = A + (size_t)(m0 + ra) * K + k0 + ca;
    const unsigned* bsrc = PB + (size_t)((k0 >> 1) + rb) * N + n0 + cb;
    *(U4*)&As[ra][ca]     = *(const U4*)asrc;
    *(U4*)&As[ra][ca + 8] = *(const U4*)(asrc + 8);
    *(U4*)&Bs[rb][cb]     = *(const U4*)bsrc;
    *(U4*)&Bs[rb][cb + 4] = *(const U4*)(bsrc + 4);
    if (k0 + 64 < K) {
      __builtin_prefetch(asrc + 64, 0, 3);
      __builtin_prefetch(PB + (size_t)(((k0 + 64) >> 1) + rb) * N + n0 + cb, 0, 3);
    }
    __syncthreads();
    #pragma unroll
    for (int ks = 0; ks < 64; ks += 32) {
      FragBF af, b0, b1;
      af.q[0] = *(const U4*)&As[am][ks + hi * 16];
      af.q[1] = *(const U4*)&As[am][ks + hi * 16 + 8];
      int kp0 = (ks >> 1) + hi * 8;
      #pragma unroll
      for (int j = 0; j < 8; ++j) {
        b0.u[j] = Bs[kp0 + j][bn];
        b1.u[j] = Bs[kp0 + j][bn + 16];
      }
      acc0 = wmma_bf16(af.h, b0.h, acc0);
      acc1 = wmma_bf16(af.h, b1.h, acc1);
    }
    __syncthreads();
  }
  int colb = n0 + nw * 32 + li;
  #pragma unroll
  for (int v = 0; v < 8; ++v) {
    int row = m0 + mw * 16 + v + 8 * hi;
    float c0 = acc0[v], c1 = acc1[v];
    if (bias) { c0 += bias[colb]; c1 += bias[colb + 16]; }
    if (flags & 1) {
      u16* C = (u16*)Cv;
      C[(size_t)row * N + colb]      = f2bf(c0);
      C[(size_t)row * N + colb + 16] = f2bf(c1);
    } else if (flags & 2) {
      float* C = (float*)Cv;
      C[(size_t)row * N + colb]      += c0;
      C[(size_t)row * N + colb + 16] += c1;
    } else {
      float* C = (float*)Cv;
      C[(size_t)row * N + colb]      = c0;
      C[(size_t)row * N + colb + 16] = c1;
    }
  }
}

// ---------------------------------------------------------------------------
// Flash attention (causal): 128 threads = 4 waves, each wave owns 16 q-rows.
// Key tiles of 32; QK^T and P*V via v_wmma_f32_16x16x32_bf16.
// ---------------------------------------------------------------------------
__global__ __launch_bounds__(128)
void k_attn(const u16* __restrict__ Q, const u16* __restrict__ Km,
            const u16* __restrict__ Vm, u16* __restrict__ O, int S, int H) {
  const int D = 1024, DH = 64;
  int bh = blockIdx.x;
  int b = bh / H, h = bh % H;
  int q0 = blockIdx.y * 64;
  int tid = threadIdx.x, lane = tid & 31, wave = tid >> 5;
  int hi = lane >> 4, li = lane & 15;
  __shared__ u16 Ks[32][72];         // keys x dims (row-major, 144B rows)
  __shared__ unsigned Vs2[16][68];   // key-pair x dims (u32 pairs, 272B rows)
  __shared__ u16 Ps[4][16][40];      // per-wave P stage (80B rows)
  size_t base = ((size_t)b * S) * D + (size_t)h * DH;
  int qrow = q0 + wave * 16 + li;

  // Q fragments: 4 coalesced b128 loads + 1/sqrt(dh) scaling during repack
  FragBF qf[2];
  {
    const unsigned* qr = (const unsigned*)(Q + base + (size_t)qrow * D);
    #pragma unroll
    for (int kk = 0; kk < 2; ++kk) {
      U4 t0 = *(const U4*)(qr + (hi * 16 + kk * 32) / 2);
      U4 t1 = *(const U4*)(qr + (hi * 16 + kk * 32 + 8) / 2);
      qf[kk].u[0] = scale_pk(t0.a, 0.125f); qf[kk].u[1] = scale_pk(t0.b, 0.125f);
      qf[kk].u[2] = scale_pk(t0.c, 0.125f); qf[kk].u[3] = scale_pk(t0.d, 0.125f);
      qf[kk].u[4] = scale_pk(t1.a, 0.125f); qf[kk].u[5] = scale_pk(t1.b, 0.125f);
      qf[kk].u[6] = scale_pk(t1.c, 0.125f); qf[kk].u[7] = scale_pk(t1.d, 0.125f);
    }
  }

  float mi[8], lsum[8];
  v8f acc[4];
  #pragma unroll
  for (int v = 0; v < 8; ++v) { mi[v] = -1e30f; lsum[v] = 0.f; }
  #pragma unroll
  for (int j = 0; j < 4; ++j)
    #pragma unroll
    for (int v = 0; v < 8; ++v) acc[j][v] = 0.f;

  int nkt = (q0 + 63) / 32 + 1;
  for (int kt = 0; kt < nkt; ++kt) {
    // K tile: rows of 32 keys x 64 dims
    {
      int r = tid >> 2, c = (tid & 3) * 16;
      const u16* ks = Km + base + (size_t)(kt * 32 + r) * D + c;
      *(U4*)&Ks[r][c]     = *(const U4*)ks;
      *(U4*)&Ks[r][c + 8] = *(const U4*)(ks + 8);
    }
    // V tile: interleave adjacent keys into u32 key-pairs in registers
    {
      int kp = tid >> 3, cv = (tid & 7) * 8;
      const u16* v0 = Vm + base + (size_t)(kt * 32 + 2 * kp) * D + cv;
      U4 a = *(const U4*)v0;
      U4 bq = *(const U4*)(v0 + D);
      U4 o0, o1;
      interleave(a.a, bq.a, o0.a, o0.b);
      interleave(a.b, bq.b, o0.c, o0.d);
      interleave(a.c, bq.c, o1.a, o1.b);
      interleave(a.d, bq.d, o1.c, o1.d);
      *(U4*)&Vs2[kp][cv]     = o0;
      *(U4*)&Vs2[kp][cv + 4] = o1;
    }
    __syncthreads();

    // S = Q K^T : two 16x16 C-frags; B-frag = 2x ds_load_b128 from K tile row
    v8f sf[2];
    #pragma unroll
    for (int f = 0; f < 2; ++f) {
      v8f s;
      #pragma unroll
      for (int v = 0; v < 8; ++v) s[v] = 0.f;
      #pragma unroll
      for (int kk = 0; kk < 2; ++kk) {
        FragBF bb;
        const u16* kr = &Ks[f * 16 + li][kk * 32 + hi * 16];
        bb.q[0] = *(const U4*)kr;
        bb.q[1] = *(const U4*)(kr + 8);
        s = wmma_bf16(qf[kk].h, bb.h, s);
      }
      sf[f] = s;
    }

    // online softmax (rows live in 16-lane halves)
    #pragma unroll
    for (int v = 0; v < 8; ++v) {
      int row = q0 + wave * 16 + v + 8 * hi;
      int key0 = kt * 32 + li, key1 = key0 + 16;
      float x0 = (key0 <= row) ? sf[0][v] : -1e30f;
      float x1 = (key1 <= row) ? sf[1][v] : -1e30f;
      float rm = fmaxf(x0, x1);
      #pragma unroll
      for (int o = 1; o < 16; o <<= 1) rm = fmaxf(rm, __shfl_xor(rm, o, 32));
      float nm = fmaxf(mi[v], rm);
      float p0 = __expf(x0 - nm), p1 = __expf(x1 - nm);
      float rs = p0 + p1;
      #pragma unroll
      for (int o = 1; o < 16; o <<= 1) rs += __shfl_xor(rs, o, 32);
      float scl = __expf(mi[v] - nm);
      lsum[v] = lsum[v] * scl + rs;
      mi[v] = nm;
      #pragma unroll
      for (int j = 0; j < 4; ++j) acc[j][v] *= scl;
      Ps[wave][v + 8 * hi][li]      = f2bf(p0);
      Ps[wave][v + 8 * hi][li + 16] = f2bf(p1);
    }
    __syncthreads();

    // P as A-frag (2x b128), V B-frags (8x b32 each), 4 wmma
    FragBF pf;
    pf.q[0] = *(const U4*)&Ps[wave][li][hi * 16];
    pf.q[1] = *(const U4*)&Ps[wave][li][hi * 16 + 8];
    #pragma unroll
    for (int j = 0; j < 4; ++j) {
      FragBF bb;
      #pragma unroll
      for (int i2 = 0; i2 < 8; ++i2)
        bb.u[i2] = Vs2[hi * 8 + i2][j * 16 + li];
      acc[j] = wmma_bf16(pf.h, bb.h, acc[j]);
    }
    __syncthreads();
  }

  #pragma unroll
  for (int j = 0; j < 4; ++j)
    #pragma unroll
    for (int v = 0; v < 8; ++v) {
      int row = q0 + wave * 16 + v + 8 * hi;
      float o = acc[j][v] / lsum[v];
      O[base + (size_t)row * D + j * 16 + li] = f2bf(o);
    }
}

// ---------------------------------------------------------------------------
// Host launcher
// ---------------------------------------------------------------------------
extern "C" void kernel_launch(void* const* d_in, const int* in_sizes, int n_in,
                              void* d_out, int out_size, void* d_ws, size_t ws_size,
                              hipStream_t stream) {
  const int Bq = 4, S = 1024, D = 1024, H = 16;
  const int T = Bq * S;
  const float* x        = (const float*)d_in[0];
  const float* qk_emb   = (const float*)d_in[1];
  const float* v_emb    = (const float*)d_in[2];
  const float* know_emb = (const float*)d_in[3];
  const float* qk_f     = (const float*)d_in[4];
  const float* qk_r     = (const float*)d_in[5];
  const float* v_f      = (const float*)d_in[6];
  const float* v_r      = (const float*)d_in[7];
  const float* know_f   = (const float*)d_in[8];
  const float* know_r   = (const float*)d_in[9];
  const float* pa_k     = (const float*)d_in[10];
  const float* pa_b     = (const float*)d_in[11];
  const float* ta_k     = (const float*)d_in[12];
  const float* ta_b     = (const float*)d_in[13];
  const float* pk_k     = (const float*)d_in[14];
  const float* pk_b     = (const float*)d_in[15];
  const float* tk_k     = (const float*)d_in[16];
  const float* tk_b     = (const float*)d_in[17];
  const float* expO     = (const float*)d_in[18];
  const float* ln1_s    = (const float*)d_in[19];
  const float* ln1_b    = (const float*)d_in[20];
  const float* ln2_s    = (const float*)d_in[21];
  const float* ln2_b    = (const float*)d_in[22];
  float* out = (float*)d_out;               // [T*D] running residual + aux scalar

  char* wsb = (char*)d_ws;
  size_t off = 0;
  auto alloc = [&](size_t bytes) -> void* {
    void* p = wsb + off;
    off += (bytes + 255) & ~(size_t)255;
    return p;
  };
  // f32 buffers
  float* nF    = (float*)alloc((size_t)T * D * 4);
  float* hall  = (float*)alloc((size_t)T * 384 * 4);
  float* tauA  = (float*)alloc((size_t)T * 3 * 4);
  float* tauK  = (float*)alloc((size_t)T * 4);
  float* score = (float*)alloc((size_t)T * 4096 * 4);   // scores / all_h (reused)
  float* gq    = (float*)alloc((size_t)T * 16 * 4);
  float* gk    = (float*)alloc((size_t)T * 16 * 4);
  float* gv    = (float*)alloc((size_t)T * 16 * 4);
  float* gkn   = (float*)alloc((size_t)T * 32 * 4);
  float* hq    = (float*)alloc((size_t)T * 128 * 4);
  float* hk    = (float*)alloc((size_t)T * 128 * 4);
  float* hv    = (float*)alloc((size_t)T * 128 * 4);
  float* hkn   = (float*)alloc((size_t)T * 128 * 4);
  float* sums  = (float*)alloc((size_t)(2048 * 3 + 4096) * 4);
  float* sumq = sums, *sumk = sums + 2048, *sumv = sums + 4096, *sumkn = sums + 6144;
  // bf16 A-operand buffers
  u16* nB    = (u16*)alloc((size_t)T * D * 2);
  u16* hqB   = (u16*)alloc((size_t)T * 128 * 2);
  u16* hkB   = (u16*)alloc((size_t)T * 128 * 2);
  u16* hvB   = (u16*)alloc((size_t)T * 128 * 2);
  u16* hknB  = (u16*)alloc((size_t)T * 128 * 2);
  u16* expB  = (u16*)alloc((size_t)T * 4096 * 2);
  u16* Qb    = (u16*)alloc((size_t)T * D * 2);
  u16* Kb    = (u16*)alloc((size_t)T * D * 2);
  u16* Vb    = (u16*)alloc((size_t)T * D * 2);
  u16* Ab    = (u16*)alloc((size_t)T * D * 2);
  // packed-B u32 weight buffers (same byte size as bf16 [K][N])
  unsigned* w_pa  = (unsigned*)alloc((size_t)512 * 384 * 4);
  unsigned* w_pk  = (unsigned*)alloc((size_t)512 * 128 * 4);
  unsigned* w_qkT = (unsigned*)alloc((size_t)64 * 2048 * 4);
  unsigned* w_vT  = (unsigned*)alloc((size_t)64 * 2048 * 4);
  unsigned* w_knT = (unsigned*)alloc((size_t)64 * 4096 * 4);
  unsigned* w_Fqk = (unsigned*)alloc((size_t)512 * 2048 * 4);
  unsigned* w_Rqk = (unsigned*)alloc((size_t)1024 * 1024 * 4);
  unsigned* w_Fv  = (unsigned*)alloc((size_t)512 * 2048 * 4);
  unsigned* w_Rv  = (unsigned*)alloc((size_t)1024 * 1024 * 4);
  unsigned* w_Fkn = (unsigned*)alloc((size_t)512 * 4096 * 4);
  unsigned* w_Rkn = (unsigned*)alloc((size_t)2048 * 1024 * 4);
  unsigned* w_O   = (unsigned*)alloc((size_t)512 * 1024 * 4);
  (void)ws_size; (void)in_sizes; (void)n_in; (void)out_size;

  auto gemm = [&](const u16* A, const unsigned* PB, void* C, const float* bias,
                  int M, int N, int K, int flags) {
    dim3 g(N / 64, M / 64);
    k_gemm<<<g, 256, 0, stream>>>(A, PB, C, bias, M, N, K, flags);
  };

  // ---- init ----
  k_copy<<<4096, 256, 0, stream>>>(out, x, T * D);
  k_zero<<<64, 256, 0, stream>>>(sums, 2048 * 3 + 4096);

  // ---- pack weights (packed-B u32 format) ----
  k_convPB<<<2048, 256, 0, stream>>>(w_pa, pa_k, 1024, 384);
  k_convPB<<<2048, 256, 0, stream>>>(w_pk, pk_k, 1024, 128);
  k_embTP<<<2048, 128, 0, stream>>>(qk_emb, w_qkT, 2048);
  k_embTP<<<2048, 128, 0, stream>>>(v_emb, w_vT, 2048);
  k_embTP<<<4096, 128, 0, stream>>>(know_emb, w_knT, 4096);
  k_packFP<<<4096, 256, 0, stream>>>(w_Fqk, qk_f, 16, 512 * 2048);
  k_convPB<<<4096, 256, 0, stream>>>(w_Rqk, qk_r, 2048, 1024);
  k_packFP<<<4096, 256, 0, stream>>>(w_Fv, v_f, 16, 512 * 2048);
  k_convPB<<<4096, 256, 0, stream>>>(w_Rv, v_r, 2048, 1024);
  k_packFP<<<4096, 256, 0, stream>>>(w_Fkn, know_f, 32, 512 * 4096);
  k_convPB<<<4096, 256, 0, stream>>>(w_Rkn, know_r, 4096, 1024);
  k_convPB<<<4096, 256, 0, stream>>>(w_O, expO, 1024, 1024);

  // ============== attention circuit ==============
  k_ln<<<T, 256, 0, stream>>>(out, ln1_s, ln1_b, nF, nB, D);
  gemm(nB, w_pa, hall, pa_b, T, 384, D, 0);
  k_tau<<<T, 256, 0, stream>>>(nF, ta_k, ta_b, tauA, D, 3);
  k_convS<<<2048, 256, 0, stream>>>(hqB, hall, T, 128, 384, 0);
  k_convS<<<2048, 256, 0, stream>>>(hkB, hall, T, 128, 384, 128);
  k_convS<<<2048, 256, 0, stream>>>(hvB, hall, T, 128, 384, 256);

  gemm(hqB, w_qkT, score, nullptr, T, 2048, 128, 0);
  k_gate<<<T, 256, 0, stream>>>(score, tauA, 3, 0, 2048, 16, sumq, gq);
  gemm(hkB, w_qkT, score, nullptr, T, 2048, 128, 0);
  k_gate<<<T, 256, 0, stream>>>(score, tauA, 3, 1, 2048, 16, sumk, gk);
  gemm(hvB, w_vT, score, nullptr, T, 2048, 128, 0);
  k_gate<<<T, 256, 0, stream>>>(score, tauA, 3, 2, 2048, 16, sumv, gv);

  gemm(nB, w_Fqk, score, nullptr, T, 2048, 1024, 0);
  k_wred<<<T, 128, 0, stream>>>(score, gq, hq, 16);
  k_wred<<<T, 128, 0, stream>>>(score, gk, hk, 16);
  k_exp<<<4096, 256, 0, stream>>>(expB, hq, gq, 16, T * 2048);
  gemm(expB, w_Rqk, Qb, nullptr, T, 1024, 2048, 1);
  k_exp<<<4096, 256, 0, stream>>>(expB, hk, gk, 16, T * 2048);
  gemm(expB, w_Rqk, Kb, nullptr, T, 1024, 2048, 1);
  gemm(nB, w_Fv, score, nullptr, T, 2048, 1024, 0);
  k_wred<<<T, 128, 0, stream>>>(score, gv, hv, 16);
  k_exp<<<4096, 256, 0, stream>>>(expB, hv, gv, 16, T * 2048);
  gemm(expB, w_Rv, Vb, nullptr, T, 1024, 2048, 1);

  dim3 ag(Bq * H, S / 64);
  k_attn<<<ag, 128, 0, stream>>>(Qb, Kb, Vb, Ab, S, H);
  gemm(Ab, w_O, out, nullptr, T, 1024, 1024, 2);         // x += attn @ O

  // ============== knowledge circuit ==============
  k_ln<<<T, 256, 0, stream>>>(out, ln2_s, ln2_b, nF, nB, D);
  gemm(nB, w_pk, hall, pk_b, T, 128, D, 0);
  k_tau<<<T, 256, 0, stream>>>(nF, tk_k, tk_b, tauK, D, 1);
  k_convS<<<2048, 256, 0, stream>>>(hknB, hall, T, 128, 128, 0);
  gemm(hknB, w_knT, score, nullptr, T, 4096, 128, 0);
  k_gate<<<T, 256, 0, stream>>>(score, tauK, 1, 0, 4096, 32, sumkn, gkn);
  gemm(nB, w_Fkn, score, nullptr, T, 4096, 1024, 0);
  k_wred<<<T, 128, 0, stream>>>(score, gkn, hkn, 32);
  k_exp<<<4096, 256, 0, stream>>>(expB, hkn, gkn, 32, T * 4096);
  gemm(expB, w_Rkn, out, nullptr, T, 1024, 4096, 2);     // x += know_out

  // ---- aux loss scalar ----
  k_aux<<<1, 256, 0, stream>>>(sumq, sumk, sumv, sumkn, 1.f / (float)T,
                               out + (size_t)T * D);
}
